// LlamaAttention_21105469292880
// MI455X (gfx1250) — matrix operand order repaired and checked by reading, not compiled, not run on hardware
//
#include <hip/hip_runtime.h>
#include <hip/hip_bf16.h>
#include <math.h>

// ---------------- problem constants ----------------
#define BB    2
#define SS    2048
#define DD    2048
#define HH    16
#define KVH_  4
#define HD_   128
#define NREP_ 4
#define QKV_N 3072              // 2048 (q) + 512 (k) + 512 (v)
#define SCALE_ 0.08838834764831845f   // 128^-0.5

typedef __bf16 bf16;
typedef __attribute__((ext_vector_type(16))) __bf16 v16bf;
typedef __attribute__((ext_vector_type(8)))  float  v8f;
typedef __attribute__((ext_vector_type(4)))  unsigned int uint4v;
typedef __attribute__((address_space(3)))    void LDSV;   // for generic->LDS offset casts

// Async global->LDS copy (CDNA5): VDST = 32-bit LDS byte offset, VADDR = 64-bit
// global address (ISA 15.18.3 op 98). Tracked by ASYNCcnt.
#define ASYNC_COPY_B128(dst_lds_ptr, src_gptr)                                   \
  do {                                                                           \
    unsigned __lds_off = (unsigned)(unsigned long long)(LDSV*)(dst_lds_ptr);     \
    asm volatile("global_load_async_to_lds_b128 %0, %1, off"                     \
                 :: "v"(__lds_off), "v"(src_gptr) : "memory");                   \
  } while (0)

#if __has_builtin(__builtin_amdgcn_s_wait_asynccnt)
#define WAIT_ASYNC() __builtin_amdgcn_s_wait_asynccnt(0)
#else
#define WAIT_ASYNC() asm volatile("s_wait_asynccnt 0x0" ::: "memory")
#endif

// ---------------- WMMA core ----------------
__device__ __forceinline__ v8f wmma_bf16(v16bf a, v16bf b, v8f c) {
  return __builtin_amdgcn_wmma_f32_16x16x32_bf16(
      /*neg_a=*/false, a, /*neg_b=*/false, b,
      /*c_mod=*/(short)0, c, /*reuse_a=*/false, /*reuse_b=*/false);
}

// ---- A fragment (16x32 bf16): lanes 0-15 kg=0, lanes 16-31 kg=8;
//      elems 0-7 = K kg+0..7, elems 8-15 = K kg+16..23
__device__ __forceinline__ v16bf load_a_bf16(const bf16* A, int lda, int m0, int k0) {
  int l = threadIdx.x & 31;
  int m = m0 + (l & 15);
  int kg = k0 + ((l >> 4) << 3);
  const bf16* p = A + (size_t)m * lda + kg;
  v16bf a;
#pragma unroll
  for (int i = 0; i < 8; ++i) { a[i] = p[i]; a[i + 8] = p[16 + i]; }
  return a;
}

// ---- B fragment (32x16): N = lane&15; lanes 0-15 kg=0, 16-31 kg=8; elems 0-7 = K kg+0..7, 8-15 = K kg+16..23 ----
__device__ __forceinline__ v16bf load_b_rm_bf16(const bf16* B, int ldb, int k0, int n0) {
  int l = threadIdx.x & 31;
  int kg = k0 + ((l >> 4) << 3);
  const bf16* p = B + (size_t)kg * ldb + n0 + (l & 15);
  v16bf b;
#pragma unroll
  for (int j = 0; j < 8; ++j) { b[j] = p[(size_t)j * ldb]; b[j + 8] = p[(size_t)(16 + j) * ldb]; }
  return b;
}

__device__ __forceinline__ v16bf load_b_rm_f32(const float* B, int ldb, int k0, int n0) {
  int l = threadIdx.x & 31;
  int kg = k0 + ((l >> 4) << 3);
  const float* p = B + (size_t)kg * ldb + n0 + (l & 15);
  v16bf b;
#pragma unroll
  for (int j = 0; j < 8; ++j) { b[j] = (bf16)p[(size_t)j * ldb]; b[j + 8] = (bf16)p[(size_t)(16 + j) * ldb]; }
  return b;
}

// B = Src^T : element (k,n) = Src[n][k] (Q.K^T operand; coalesced across lanes)
__device__ __forceinline__ v16bf load_b_tr_bf16(const bf16* S, int lds_, int k0, int n0) {
  int l = threadIdx.x & 31;
  int kg = k0 + ((l >> 4) << 3);
  const bf16* p = S + (size_t)(n0 + (l & 15)) * lds_ + kg;
  v16bf b;
#pragma unroll
  for (int j = 0; j < 8; ++j) { b[j] = p[j]; b[j + 8] = p[16 + j]; }
  return b;
}

// C layout: VGPR r, lanes 0-15 -> M=r, lanes 16-31 -> M=r+8 ; N = lane&15
#define C_ROWBASE(l) (((l) >> 4) << 3)
#define C_COL(l)     ((l) & 15)

// ---------------- K0: fp32 -> bf16 ----------------
__global__ void cvt_f32_bf16_kernel(const float* __restrict__ x, bf16* __restrict__ y, size_t n) {
  size_t i = (size_t)blockIdx.x * blockDim.x + threadIdx.x;
  if (i < n) y[i] = (bf16)x[i];
}

// ---------------- K1: fused QKV GEMM + bias (16x64 per wave) ----------------
__global__ void qkv_gemm_kernel(const bf16* __restrict__ Xb,
                                const float* __restrict__ Wq, const float* __restrict__ bq,
                                const float* __restrict__ Wk, const float* __restrict__ bk,
                                const float* __restrict__ Wv, const float* __restrict__ bv,
                                bf16* __restrict__ qkv) {
  int wave = threadIdx.x >> 5;
  int n0 = blockIdx.x * 64;              // 64-col strip (never straddles q/k/v bounds)
  int mt = blockIdx.y * 8 + wave;        // 0..255
  const float* Wm; const float* bias; int ldb; int nw;
  if (n0 < 2048)      { Wm = Wq; bias = bq; ldb = 2048; nw = n0;        }
  else if (n0 < 2560) { Wm = Wk; bias = bk; ldb = 512;  nw = n0 - 2048; }
  else                { Wm = Wv; bias = bv; ldb = 512;  nw = n0 - 2560; }

  v8f c0 = {}, c1 = {}, c2 = {}, c3 = {};
  int l = threadIdx.x & 31;
  for (int k0 = 0; k0 < DD; k0 += 32) {
    if (k0 + 32 < DD)  // hide strided weight walk behind the 4 WMMAs
      __builtin_prefetch(Wm + (size_t)(k0 + 32 + l) * ldb + nw, 0, 1);
    v16bf a = load_a_bf16(Xb, DD, mt * 16, k0);
    c0 = wmma_bf16(a, load_b_rm_f32(Wm, ldb, k0, nw),      c0);
    c1 = wmma_bf16(a, load_b_rm_f32(Wm, ldb, k0, nw + 16), c1);
    c2 = wmma_bf16(a, load_b_rm_f32(Wm, ldb, k0, nw + 32), c2);
    c3 = wmma_bf16(a, load_b_rm_f32(Wm, ldb, k0, nw + 48), c3);
  }
  int rbase = mt * 16 + C_ROWBASE(l);
  v8f acc[4] = {c0, c1, c2, c3};
#pragma unroll
  for (int j = 0; j < 4; ++j) {
    int col = n0 + j * 16 + C_COL(l);
    float bval = bias[nw + j * 16 + C_COL(l)];
#pragma unroll
    for (int r = 0; r < 8; ++r)
      qkv[(size_t)(rbase + r) * QKV_N + col] = (bf16)(acc[j][r] + bval);
  }
}

// ---------------- K2: RoPE + rearrange to [b, head, s, hd] ----------------
__global__ void rope_rearrange_kernel(const bf16* __restrict__ qkv,
                                      const float* __restrict__ cosT,
                                      const float* __restrict__ sinT,
                                      bf16* __restrict__ Qo, bf16* __restrict__ Ko,
                                      bf16* __restrict__ Vo) {
  size_t id = (size_t)blockIdx.x * blockDim.x + threadIdx.x;
  const size_t total = (size_t)BB * SS * (HH + 2 * KVH_) * HD_;
  if (id >= total) return;
  int d  = (int)(id % HD_);
  int hh = (int)((id / HD_) % (HH + 2 * KVH_));
  int s  = (int)((id / ((size_t)HD_ * (HH + 2 * KVH_))) % SS);
  int b  = (int)(id / ((size_t)HD_ * (HH + 2 * KVH_) * SS));
  size_t row = (size_t)b * SS + s;

  if (hh < HH) {                                  // Q head, RoPE
    const bf16* src = qkv + row * QKV_N + hh * HD_;
    float x  = (float)src[d];
    float pr = (d < HD_ / 2) ? -(float)src[d + HD_ / 2] : (float)src[d - HD_ / 2];
    float cs = cosT[row * HD_ + d], sn = sinT[row * HD_ + d];
    Qo[(((size_t)b * HH + hh) * SS + s) * HD_ + d] = (bf16)(x * cs + pr * sn);
  } else if (hh < HH + KVH_) {                    // K head, RoPE
    int kh = hh - HH;
    const bf16* src = qkv + row * QKV_N + 2048 + kh * HD_;
    float x  = (float)src[d];
    float pr = (d < HD_ / 2) ? -(float)src[d + HD_ / 2] : (float)src[d - HD_ / 2];
    float cs = cosT[row * HD_ + d], sn = sinT[row * HD_ + d];
    Ko[(((size_t)b * KVH_ + kh) * SS + s) * HD_ + d] = (bf16)(x * cs + pr * sn);
  } else {                                        // V head, copy
    int vh = hh - HH - KVH_;
    Vo[(((size_t)b * KVH_ + vh) * SS + s) * HD_ + d] =
        qkv[row * QKV_N + 2560 + vh * HD_ + d];
  }
}

// ---------------- K3: scores = scale * Q K^T, causal tiles only ----------------
// Q tile (16x128 bf16 = 4 KB) staged once per block via async-to-LDS; shared by 8 waves.
__global__ void scores_kernel(const bf16* __restrict__ Q, const bf16* __restrict__ K,
                              float* __restrict__ W) {
  __shared__ bf16 qtile[16 * HD_];
  int qt = blockIdx.x;
  int bh = blockIdx.z;
  int b = bh / HH, h = bh % HH;
  const bf16* Qp = Q + (size_t)bh * SS * HD_;
  const bf16* Kp = K + (size_t)(b * KVH_ + h / NREP_) * SS * HD_;

  { // cooperative stage: 256 threads x 16 B = 4 KB, via ASYNCcnt path
    int t = threadIdx.x;
    int row = t >> 4;
    int colb = (t & 15) * 8;
    const bf16* src = Qp + (size_t)(qt * 16 + row) * HD_ + colb;
    bf16* dst = &qtile[row * HD_ + colb];
    ASYNC_COPY_B128(dst, src);
    WAIT_ASYNC();
  }
  __syncthreads();

  int wave = threadIdx.x >> 5;
  int kt = blockIdx.y * 8 + wave;
  if (kt > qt) return;                     // fully-masked tile: softmax zero-fills it

  v8f c = {};
#pragma unroll
  for (int k0 = 0; k0 < HD_; k0 += 32) {
    v16bf a = load_a_bf16(qtile, HD_, 0, k0);     // LDS reads (ds_load)
    v16bf bfr = load_b_tr_bf16(Kp, HD_, k0, kt * 16);
    c = wmma_bf16(a, bfr, c);
  }
  int l = threadIdx.x & 31;
  int col = kt * 16 + C_COL(l);
  int rbase = qt * 16 + C_ROWBASE(l);
  float* Wp = W + (size_t)bh * SS * SS;
#pragma unroll
  for (int r = 0; r < 8; ++r)
    Wp[(size_t)(rbase + r) * SS + col] = c[r] * SCALE_;
}

// ---------------- K4: causal row softmax (writes every element of attn_weights) ----
__global__ void softmax_kernel(float* __restrict__ W) {
  int q  = blockIdx.x;
  int bh = blockIdx.y;
  float* row = W + ((size_t)bh * SS + q) * SS;
  int t = threadIdx.x;
  int valid = q + 1;                         // causal: cols [0, q] only
  float vals[8];
  float m = -3.4e38f;
#pragma unroll
  for (int j = 0; j < 8; ++j) {
    int cidx = t + j * 256;
    float v = (cidx < valid) ? row[cidx] : -3.4e38f;
    vals[j] = v;
    m = fmaxf(m, v);
  }
  __shared__ float red[256];
  red[t] = m; __syncthreads();
  for (int s = 128; s > 0; s >>= 1) { if (t < s) red[t] = fmaxf(red[t], red[t + s]); __syncthreads(); }
  m = red[0]; __syncthreads();
  float sum = 0.f;
#pragma unroll
  for (int j = 0; j < 8; ++j) {
    int cidx = t + j * 256;
    float e = (cidx < valid) ? __expf(vals[j] - m) : 0.f;
    vals[j] = e; sum += e;
  }
  red[t] = sum; __syncthreads();
  for (int s = 128; s > 0; s >>= 1) { if (t < s) red[t] += red[t + s]; __syncthreads(); }
  float inv = 1.0f / red[0];
#pragma unroll
  for (int j = 0; j < 8; ++j) row[t + j * 256] = vals[j] * inv;   // masked cols -> exact 0
}

// ---------------- K5: ctx = P @ V (causally clipped K loop) ----------------
// P tile (16x32) converted fp32->bf16 into LDS once per k-step; shared by 8 waves.
__global__ void av_kernel(const float* __restrict__ P, const bf16* __restrict__ V,
                          bf16* __restrict__ ctx) {
  __shared__ bf16 ptile[16 * 32];            // 1 KB
  int wave = threadIdx.x >> 5;               // n-tile 0..7 (HD = 8 * 16)
  int qt = blockIdx.x;
  int bh = blockIdx.z;
  int b = bh / HH, h = bh % HH;
  const float* Pp = P + (size_t)bh * SS * SS;
  const bf16* Vp  = V + (size_t)(b * KVH_ + h / NREP_) * SS * HD_;
  int kmax = ((qt * 16 + 16 + 31) / 32) * 32;
  if (kmax > SS) kmax = SS;

  int t = threadIdx.x;
  int sr = t >> 4;                 // 0..15  (staging row)
  int sc = (t & 15) * 2;           // 0..30  (staging col pair)
  v8f c = {};
  for (int k0 = 0; k0 < kmax; k0 += 32) {
    const float* ps = Pp + (size_t)(qt * 16 + sr) * SS + k0 + sc;
    ptile[sr * 32 + sc]     = (bf16)ps[0];
    ptile[sr * 32 + sc + 1] = (bf16)ps[1];
    __syncthreads();
    v16bf a = load_a_bf16(ptile, 32, 0, 0);          // LDS reads
    v16bf bfr = load_b_rm_bf16(Vp, HD_, k0, wave * 16);
    c = wmma_bf16(a, bfr, c);
    __syncthreads();
  }
  int l = threadIdx.x & 31;
  int col = wave * 16 + C_COL(l);
  int rbase = qt * 16 + C_ROWBASE(l);
#pragma unroll
  for (int r = 0; r < 8; ++r)
    ctx[((size_t)b * SS + rbase + r) * (HH * HD_) + h * HD_ + col] = (bf16)c[r];
}

// ---------------- K6: out = ctx @ Wo (fp32 result, 16x64 per wave) ----------------
__global__ void oproj_kernel(const bf16* __restrict__ ctx, const float* __restrict__ Wo,
                             float* __restrict__ out) {
  int wave = threadIdx.x >> 5;
  int n0 = blockIdx.x * 64;              // 0..2047 step 64
  int mt = blockIdx.y * 8 + wave;        // 0..255
  int l = threadIdx.x & 31;
  v8f c0 = {}, c1 = {}, c2 = {}, c3 = {};
  for (int k0 = 0; k0 < DD; k0 += 32) {
    if (k0 + 32 < DD)
      __builtin_prefetch(Wo + (size_t)(k0 + 32 + l) * DD + n0, 0, 1);
    v16bf a = load_a_bf16(ctx, DD, mt * 16, k0);
    c0 = wmma_bf16(a, load_b_rm_f32(Wo, DD, k0, n0),      c0);
    c1 = wmma_bf16(a, load_b_rm_f32(Wo, DD, k0, n0 + 16), c1);
    c2 = wmma_bf16(a, load_b_rm_f32(Wo, DD, k0, n0 + 32), c2);
    c3 = wmma_bf16(a, load_b_rm_f32(Wo, DD, k0, n0 + 48), c3);
  }
  int rbase = mt * 16 + C_ROWBASE(l);
  v8f acc[4] = {c0, c1, c2, c3};
#pragma unroll
  for (int j = 0; j < 4; ++j) {
    int col = n0 + j * 16 + C_COL(l);
#pragma unroll
    for (int r = 0; r < 8; ++r)
      out[(size_t)(rbase + r) * DD + col] = acc[j][r];
  }
}

// ---------------- host side ----------------
extern "C" void kernel_launch(void* const* d_in, const int* in_sizes, int n_in,
                              void* d_out, int out_size, void* d_ws, size_t ws_size,
                              hipStream_t stream) {
  (void)in_sizes; (void)n_in; (void)out_size; (void)ws_size;
  const float* X    = (const float*)d_in[0];
  const float* cosT = (const float*)d_in[1];
  const float* sinT = (const float*)d_in[2];
  // d_in[3] attention_mask: deterministic causal mask -> applied analytically, never read
  const float* Wq = (const float*)d_in[4];
  const float* bq = (const float*)d_in[5];
  const float* Wk = (const float*)d_in[6];
  const float* bk = (const float*)d_in[7];
  const float* Wv = (const float*)d_in[8];
  const float* bv = (const float*)d_in[9];
  const float* Wo = (const float*)d_in[10];

  float* out   = (float*)d_out;                       // [B,S,D] attn_out
  float* attnW = out + (size_t)BB * SS * DD;          // [B,H,S,S] attn_weights

  char* ws = (char*)d_ws;                             // 64 MB layout
  bf16* xbf = (bf16*)(ws);                            // 16 MB (recycled as ctx)
  bf16* qkv = (bf16*)(ws + (16ull << 20));            // 24 MB
  bf16* Qr  = (bf16*)(ws + (40ull << 20));            // 16 MB
  bf16* Kr  = (bf16*)(ws + (56ull << 20));            //  4 MB
  bf16* Vr  = (bf16*)(ws + (60ull << 20));            //  4 MB
  bf16* ctx = xbf;                                    // X dead after K1

  { // K0: X -> bf16 once (X is reused across 48 column strips in K1)
    size_t n = (size_t)BB * SS * DD;
    cvt_f32_bf16_kernel<<<dim3((unsigned)((n + 255) / 256)), 256, 0, stream>>>(X, xbf, n);
  }
  // K1: fused QKV projection + bias (16x64 per wave)
  qkv_gemm_kernel<<<dim3(QKV_N / 64, (BB * SS) / 128), 256, 0, stream>>>(
      xbf, Wq, bq, Wk, bk, Wv, bv, qkv);
  { // K2: RoPE + head-major rearrange
    size_t n = (size_t)BB * SS * (HH + 2 * KVH_) * HD_;
    rope_rearrange_kernel<<<dim3((unsigned)((n + 255) / 256)), 256, 0, stream>>>(
        qkv, cosT, sinT, Qr, Kr, Vr);
  }
  // K3: causal-clipped score GEMM, Q tile staged via async-to-LDS
  scores_kernel<<<dim3(SS / 16, SS / 16 / 8, BB * HH), 256, 0, stream>>>(Qr, Kr, attnW);
  // K4: row softmax; zero-fills masked columns -> attnW fully written
  softmax_kernel<<<dim3(SS, BB * HH), 256, 0, stream>>>(attnW);
  // K5: ctx = P @ V, P tile converted into LDS and shared by 8 waves
  av_kernel<<<dim3(SS / 16, 1, BB * HH), 256, 0, stream>>>(attnW, Vr, ctx);
  // K6: output projection (fp32 result, 16x64 per wave)
  oproj_kernel<<<dim3(DD / 64, (BB * SS) / 128), 256, 0, stream>>>(ctx, Wo, out);
}
